// ManualGRU_87505663689387
// MI455X (gfx1250) — compile-verified, matrix operand order checked by reading, not run on hardware
//
#include <hip/hip_runtime.h>

// ---------------------------------------------------------------------------
// GRU on MI455X (gfx1250): bf16 WMMA, f32 accumulate, persistent recurrence.
//   K0 pack_weights : f32 weights -> bf16 WMMA B-fragments in ws
//   K1 xproj        : x @ W_g[:, :I]^T + b_g  (parallel GEMM, WMMA),
//                     results stored in C-fragment layout (2xb128 per lane)
//   K2 gru_recurrent: single 1024-thread workgroup, S sequential steps,
//                     h/z/rh in LDS; emits h_s as bf16 A-fragments to ws
//   K3 out_proj     : hs-fragments (global) @ W_fc^T + b_fc -> d_out, no LDS
// ---------------------------------------------------------------------------

typedef __attribute__((ext_vector_type(16))) __bf16 v16bf;
typedef __attribute__((ext_vector_type(8)))  float  v8f;

#define B_  64
#define S_  2048
#define I_  256
#define H_  256
#define O_  256

#define KT_ 8      // 256 / 32
#define NT_ 16     // 256 / 16
#define WELEM 65536

// 16-bit A-matrix 16x32 per-lane K mapping (ISA 7.12.2), e = bf16 idx in the
// lane's 32-byte fragment:
//   lanes 0-15 : K = {0..7} u {16..23};  lanes 16-31: +8
__device__ __forceinline__ int kmap(int lane, int e) {
    int v = e >> 1, hb = e & 1;
    int base = (lane & 16) ? 8 : 0;
    return (v < 4) ? (2 * v + hb + base) : (16 + 2 * (v - 4) + hb + base);
}

__device__ __forceinline__ float sigmoidf_(float x) {
    return 1.0f / (1.0f + __expf(-x));
}
__device__ __forceinline__ float tanhf_(float x) {
    float e2 = __expf(2.0f * x);
    return (e2 - 1.0f) / (e2 + 1.0f);
}

// ---------------------------------------------------------------------------
// K0: pack 7 weight slices (0 Wr_x, 1 Wr_h, 2 Wz_x, 3 Wz_h, 4 Wh_x, 5 Wh_h,
// 6 Wfc) into bf16 B-fragment layout:
//   packed idx = w*WELEM + ((kt*16+nt)*32 + lane)*16 + e ;  B(k,n) = W[n,kofs+k]
// ---------------------------------------------------------------------------
__global__ void __launch_bounds__(256)
pack_weights(const float* __restrict__ Wr, const float* __restrict__ Wz,
             const float* __restrict__ Wh, const float* __restrict__ Wfc,
             __bf16* __restrict__ wp) {
    int idx = blockIdx.x * 256 + threadIdx.x;
    if (idx >= 7 * WELEM) return;
    int w    = idx >> 16;
    int r    = idx & (WELEM - 1);
    int e    = r & 15;
    int lane = (r >> 4) & 31;
    int tile = r >> 9;            // kt*16 + nt
    int nt   = tile & 15;
    int kt   = tile >> 4;
    int n    = nt * 16 + (lane & 15);
    int k    = kt * 32 + kmap(lane, e);

    const float* src; int indim; int kofs;
    switch (w) {
        case 0: src = Wr;  indim = I_ + H_; kofs = 0;   break;
        case 1: src = Wr;  indim = I_ + H_; kofs = I_;  break;
        case 2: src = Wz;  indim = I_ + H_; kofs = 0;   break;
        case 3: src = Wz;  indim = I_ + H_; kofs = I_;  break;
        case 4: src = Wh;  indim = I_ + H_; kofs = 0;   break;
        case 5: src = Wh;  indim = I_ + H_; kofs = I_;  break;
        default: src = Wfc; indim = H_;     kofs = 0;   break;
    }
    wp[idx] = (__bf16)src[(size_t)n * indim + kofs + k];
}

// ---------------------------------------------------------------------------
// K1: x-projection. grid = (S, 3 gates), 256 threads (8 waves).
// xpart layout (C-fragment tiles): ((s*64 + tile)*32 + lane)*8 + v  (f32),
// tile = mt*16 + nt, bias folded in.
// ---------------------------------------------------------------------------
__global__ void __launch_bounds__(256)
xproj_kernel(const float* __restrict__ x,
             const float* __restrict__ b_r, const float* __restrict__ b_z,
             const float* __restrict__ b_h,
             const __bf16* __restrict__ wp,
             float* __restrict__ xr, float* __restrict__ xz,
             float* __restrict__ xh) {
    __shared__ __attribute__((aligned(32))) __bf16 frags[4][KT_][32][16];

    int s   = blockIdx.x;
    int g   = blockIdx.y;
    int tid = threadIdx.x;

    // stage A-fragments: 1024 lane-entries, 4 per thread (f32 x -> bf16)
    for (int ei = 0; ei < 4; ++ei) {
        int ent = tid * 4 + ei;
        int ln  = ent & 31;
        int kt  = (ent >> 5) & 7;
        int mt  = ent >> 8;
        int b   = mt * 16 + (ln & 15);
        const float* row = x + ((size_t)b * S_ + s) * I_ + kt * 32;
        __bf16* dst = &frags[mt][kt][ln][0];
        #pragma unroll
        for (int e = 0; e < 16; ++e) dst[e] = (__bf16)row[kmap(ln, e)];
    }
    __syncthreads();

    int wave = tid >> 5, lane = tid & 31;
    const float* bias = (g == 0) ? b_r : (g == 1) ? b_z : b_h;
    float* outp       = (g == 0) ? xr  : (g == 1) ? xz  : xh;
    const __bf16* wb  = wp + (size_t)(g * 2) * WELEM;   // x-slices: 0,2,4

    for (int q = 0; q < 8; ++q) {
        int job = wave * 8 + q;          // tile id = mt*16 + nt
        int mt = job >> 4, nt = job & 15;
        int n  = nt * 16 + (lane & 15);
        float bv = bias[n];
        v8f acc;
        #pragma unroll
        for (int v = 0; v < 8; ++v) acc[v] = bv;

        v16bf barr[KT_];                 // batch all B fragments (one clause)
        #pragma unroll
        for (int kt = 0; kt < KT_; ++kt)
            barr[kt] = *(const v16bf*)(wb + (((kt * 16) + nt) * 32 + lane) * 16);
        #pragma unroll
        for (int kt = 0; kt < KT_; ++kt) {
            v16bf a = *(const v16bf*)&frags[mt][kt][lane][0];
            acc = __builtin_amdgcn_wmma_f32_16x16x32_bf16(
                      false, a, false, barr[kt], (short)0, acc, false, false);
        }

        float4* op = (float4*)(outp + (((size_t)s * 64 + job) * 32 + lane) * 8);
        float4 s0 = {acc[0], acc[1], acc[2], acc[3]};
        float4 s1 = {acc[4], acc[5], acc[6], acc[7]};
        op[0] = s0;
        op[1] = s1;
    }
}

// ---------------------------------------------------------------------------
// K2: sequential recurrence. ONE workgroup, 1024 threads = 32 waves.
// LDS: h f32 (64KB) + fragment buffer (32KB) + r*h bf16 (32KB) + z bf16 (32KB)
// Emits h_s as bf16 A-fragments into hsf; h_last (f32) into d_out tail.
// hsf layout: (((s*4 + mt)*8 + kt)*32 + lane)*16 + e
// ---------------------------------------------------------------------------
__global__ void __launch_bounds__(1024)
gru_recurrent(const __bf16* __restrict__ wp,
              const float* __restrict__ xr, const float* __restrict__ xz,
              const float* __restrict__ xh,
              __bf16* __restrict__ hsf,
              float* __restrict__ out) {
    __shared__ float h_f32[B_][H_];                                       // 64 KB
    __shared__ __attribute__((aligned(32))) __bf16 frags[4][KT_][32][16]; // 32 KB
    __shared__ __bf16 rh_st[B_][H_];                                      // 32 KB
    __shared__ __bf16 z_st[B_][H_];                                       // 32 KB

    int tid = threadIdx.x, wave = tid >> 5, lane = tid & 31;

    // warm packed weights into cache (global_prefetch_b8)
    {
        const char* wb = (const char*)wp;
        for (size_t o = (size_t)tid * 128; o < (size_t)7 * WELEM * 2;
             o += 1024 * 128)
            __builtin_prefetch(wb + o, 0, 3);
    }

    for (int i = tid; i < B_ * H_; i += 1024) ((float*)h_f32)[i] = 0.0f;
    __syncthreads();

    const __bf16* Wrh = wp + (size_t)1 * WELEM;
    const __bf16* Wzh = wp + (size_t)3 * WELEM;
    const __bf16* Whh = wp + (size_t)5 * WELEM;

    #pragma unroll 1
    for (int s = 0; s < S_; ++s) {
        // Phase A: build h A-fragments (one lane-entry per thread)
        {
            int ln = tid & 31, kt = (tid >> 5) & 7, mt = tid >> 8;
            int m  = mt * 16 + (ln & 15);
            __bf16* dst = &frags[mt][kt][ln][0];
            #pragma unroll
            for (int e = 0; e < 16; ++e)
                dst[e] = (__bf16)h_f32[m][kt * 32 + kmap(ln, e)];
        }
        __syncthreads();

        // Phase B: r and z gates (128 tile jobs, 4 per wave)
        #pragma unroll 1
        for (int q = 0; q < 4; ++q) {
            int job  = wave * 4 + q;
            int gate = job >> 6;         // 0 = r, 1 = z
            int rem  = job & 63;         // tile id = mt*16 + nt
            int mt = rem >> 4, nt = rem & 15;
            int n  = nt * 16 + (lane & 15);
            const float* xg  = gate ? xz  : xr;
            const __bf16* wb = gate ? Wzh : Wrh;

            v16bf barr[KT_];
            #pragma unroll
            for (int kt = 0; kt < KT_; ++kt)
                barr[kt] = *(const v16bf*)(wb + ((kt * 16 + nt) * 32 + lane) * 16);

            const float4* ap =
                (const float4*)(xg + (((size_t)s * 64 + rem) * 32 + lane) * 8);
            float4 a0 = ap[0], a1 = ap[1];
            v8f acc;
            acc[0] = a0.x; acc[1] = a0.y; acc[2] = a0.z; acc[3] = a0.w;
            acc[4] = a1.x; acc[5] = a1.y; acc[6] = a1.z; acc[7] = a1.w;

            #pragma unroll
            for (int kt = 0; kt < KT_; ++kt) {
                v16bf a = *(const v16bf*)&frags[mt][kt][lane][0];
                acc = __builtin_amdgcn_wmma_f32_16x16x32_bf16(
                          false, a, false, barr[kt], (short)0, acc, false, false);
            }
            #pragma unroll
            for (int v = 0; v < 8; ++v) {
                int m = mt * 16 + v + ((lane >> 4) << 3);
                float sg = sigmoidf_(acc[v]);
                if (gate == 0) rh_st[m][n] = (__bf16)(sg * h_f32[m][n]);
                else           z_st[m][n]  = (__bf16)sg;
            }
        }
        __syncthreads();

        // Phase C: build (r*h) A-fragments
        {
            int ln = tid & 31, kt = (tid >> 5) & 7, mt = tid >> 8;
            int m  = mt * 16 + (ln & 15);
            __bf16* dst = &frags[mt][kt][ln][0];
            #pragma unroll
            for (int e = 0; e < 16; ++e)
                dst[e] = rh_st[m][kt * 32 + kmap(ln, e)];
        }
        __syncthreads();

        // Phase D: candidate h_tilde + state update (64 jobs, 2 per wave)
        #pragma unroll 1
        for (int q = 0; q < 2; ++q) {
            int job = wave * 2 + q;      // tile id = mt*16 + nt
            int mt = job >> 4, nt = job & 15;
            int n  = nt * 16 + (lane & 15);

            v16bf barr[KT_];
            #pragma unroll
            for (int kt = 0; kt < KT_; ++kt)
                barr[kt] = *(const v16bf*)(Whh + ((kt * 16 + nt) * 32 + lane) * 16);

            const float4* ap =
                (const float4*)(xh + (((size_t)s * 64 + job) * 32 + lane) * 8);
            float4 a0 = ap[0], a1 = ap[1];
            v8f acc;
            acc[0] = a0.x; acc[1] = a0.y; acc[2] = a0.z; acc[3] = a0.w;
            acc[4] = a1.x; acc[5] = a1.y; acc[6] = a1.z; acc[7] = a1.w;

            #pragma unroll
            for (int kt = 0; kt < KT_; ++kt) {
                v16bf a = *(const v16bf*)&frags[mt][kt][lane][0];
                acc = __builtin_amdgcn_wmma_f32_16x16x32_bf16(
                          false, a, false, barr[kt], (short)0, acc, false, false);
            }

            // destination within the hs A-fragment tile for column n
            int kt2     = n >> 5;
            int kcol    = n & 31;
            int lane_hi = (kcol >> 3) & 1;          // k in {8..15,24..31} -> hi lanes
            int kk      = kcol - 8 * lane_hi;       // in {0..7} u {16..23}
            int e2      = (kk & 7) + ((kk & 16) ? 8 : 0);

            #pragma unroll
            for (int v = 0; v < 8; ++v) {
                int m  = mt * 16 + v + ((lane >> 4) << 3);
                float ht = tanhf_(acc[v]);
                float z  = (float)z_st[m][n];
                float hn = (1.0f - z) * h_f32[m][n] + z * ht;
                h_f32[m][n] = hn;
                int lane2 = lane_hi * 16 + (m & 15);
                hsf[((((size_t)s * 4 + mt) * 8 + kt2) * 32 + lane2) * 16 + e2] =
                    (__bf16)hn;
                if (s == S_ - 1)
                    out[(size_t)B_ * S_ * O_ + (size_t)m * H_ + n] = hn; // h_last
            }
        }
        __syncthreads();
    }
}

// ---------------------------------------------------------------------------
// K3: output projection from hs fragments. grid = S blocks, 256 threads.
// Pure register GEMM: A frags + B frags from global, WMMA, store to d_out.
// ---------------------------------------------------------------------------
__global__ void __launch_bounds__(256)
out_proj(const float* __restrict__ b_fc, const __bf16* __restrict__ wp,
         const __bf16* __restrict__ hsf, float* __restrict__ out) {
    int s    = blockIdx.x;
    int tid  = threadIdx.x;
    int wave = tid >> 5, lane = tid & 31;
    const __bf16* Wfc = wp + (size_t)6 * WELEM;

    for (int q = 0; q < 8; ++q) {
        int job = wave * 8 + q;          // 0..63 : mt*16 + nt
        int mt = job >> 4, nt = job & 15;
        int n  = nt * 16 + (lane & 15);
        float bv = b_fc[n];
        v8f acc;
        #pragma unroll
        for (int v = 0; v < 8; ++v) acc[v] = bv;

        v16bf barr[KT_];
        #pragma unroll
        for (int kt = 0; kt < KT_; ++kt)
            barr[kt] = *(const v16bf*)(Wfc + ((kt * 16 + nt) * 32 + lane) * 16);
        v16bf aarr[KT_];
        #pragma unroll
        for (int kt = 0; kt < KT_; ++kt)
            aarr[kt] = *(const v16bf*)(hsf +
                       ((((size_t)s * 4 + mt) * 8 + kt) * 32 + lane) * 16);
        #pragma unroll
        for (int kt = 0; kt < KT_; ++kt)
            acc = __builtin_amdgcn_wmma_f32_16x16x32_bf16(
                      false, aarr[kt], false, barr[kt], (short)0, acc, false, false);

        #pragma unroll
        for (int v = 0; v < 8; ++v) {
            int b = mt * 16 + v + ((lane >> 4) << 3);     // batch row
            out[((size_t)b * S_ + s) * O_ + n] = acc[v];
        }
    }
}

// ---------------------------------------------------------------------------
extern "C" void kernel_launch(void* const* d_in, const int* in_sizes, int n_in,
                              void* d_out, int out_size, void* d_ws, size_t ws_size,
                              hipStream_t stream) {
    const float* x    = (const float*)d_in[0];
    const float* W_r  = (const float*)d_in[1];
    const float* b_r  = (const float*)d_in[2];
    const float* W_z  = (const float*)d_in[3];
    const float* b_z  = (const float*)d_in[4];
    const float* W_h  = (const float*)d_in[5];
    const float* b_h  = (const float*)d_in[6];
    const float* W_fc = (const float*)d_in[7];
    const float* b_fc = (const float*)d_in[8];
    float* out = (float*)d_out;

    // ws layout: [0,1MB) packed bf16 weights; 3 x 128MB f32 x-projections;
    //            64MB bf16 hs fragments.
    __bf16* wp = (__bf16*)d_ws;
    const size_t GATE_ELEMS = (size_t)S_ * 64 * 256;
    float* xr = (float*)((char*)d_ws + (size_t)(1u << 20));
    float* xz = xr + GATE_ELEMS;
    float* xh = xz + GATE_ELEMS;
    __bf16* hsf = (__bf16*)(xh + GATE_ELEMS);

    pack_weights<<<(7 * WELEM) / 256, 256, 0, stream>>>(W_r, W_z, W_h, W_fc, wp);
    xproj_kernel<<<dim3(S_, 3), 256, 0, stream>>>(x, b_r, b_z, b_h, wp, xr, xz, xh);
    gru_recurrent<<<1, 1024, 0, stream>>>(wp, xr, xz, xh, hsf, out);
    out_proj<<<S_, 256, 0, stream>>>(b_fc, wp, hsf, out);
}
